// GraphIsomorphismNetwork_59571196396107
// MI455X (gfx1250) — matrix operand order repaired and checked by reading, not compiled, not run on hardware
//
#include <hip/hip_runtime.h>

// ---------------------------------------------------------------------------
// GIN forward for MI455X (gfx1250), wave32 + WMMA bf16.
#define NNODES 200000
#define NEDGES 400000
#define NGRAPH 4096
#define HDIM   300
#define HP     304      // H padded to 16
#define H2     600
#define ZP     608      // 2H padded to 16 (and to 32 for GEMM2 K: 608 = 19*32)
#define KP1    320      // GEMM1 K (H=300) padded to 32
#define NLAYER 5
#define NT1    38       // GEMM1 output N-tiles (608/16)
#define NT2    19       // GEMM2 output N-tiles (304/16)

typedef __attribute__((ext_vector_type(16))) __bf16 v16bf;
typedef __attribute__((ext_vector_type(8)))  float  v8f;

union Frag { v16bf v; unsigned u[8]; };

__device__ __forceinline__ unsigned short f2bf(float f) {
    unsigned u = __builtin_bit_cast(unsigned, f);
    unsigned r = u + 0x7FFFu + ((u >> 16) & 1u);   // round-to-nearest-even
    return (unsigned short)(r >> 16);
}

// ---------------------------------------------------------------------------
// Convert + transpose + zero-pad weights to bf16 once per launch.
// W1t: [L][608][320]  (n-major so B-fragment k-pairs are contiguous b32 loads)
// W2t: [L][304][608]
__global__ void prep_weights(const float* __restrict__ W1, const float* __restrict__ W2,
                             unsigned short* __restrict__ W1t, unsigned short* __restrict__ W2t) {
    const int n1 = NLAYER * ZP * KP1;
    const int n2 = NLAYER * HP * ZP;
    for (int i = blockIdx.x * blockDim.x + threadIdx.x; i < n1 + n2;
         i += gridDim.x * blockDim.x) {
        if (i < n1) {
            int l = i / (ZP * KP1); int r = i % (ZP * KP1);
            int n = r / KP1;        int k = r % KP1;
            float v = (k < HDIM && n < H2) ? W1[((size_t)l * HDIM + k) * H2 + n] : 0.0f;
            W1t[i] = f2bf(v);
        } else {
            int j = i - n1;
            int l = j / (HP * ZP);  int r = j % (HP * ZP);
            int n = r / ZP;         int k = r % ZP;
            float v = (k < H2 && n < HDIM) ? W2[((size_t)l * H2 + k) * HDIM + n] : 0.0f;
            W2t[j] = f2bf(v);
        }
    }
}

// ---------------------------------------------------------------------------
// AtomEncoder: h[n][c] = sum_f atom_emb[f][x[n][f]][c]; pad cols zeroed.
__global__ void atom_enc(const int* __restrict__ x, const float* __restrict__ aemb,
                         float* __restrict__ h) {
    int n = blockIdx.x, c = threadIdx.x;
    if (c >= HP) return;
    float s = 0.0f;
    if (c < HDIM) {
        #pragma unroll
        for (int f = 0; f < 9; ++f) {
            int xv = x[n * 9 + f];
            s += aemb[((size_t)f * 120 + xv) * HDIM + c];
        }
    }
    h[(size_t)n * HP + c] = s;
}

// ---------------------------------------------------------------------------
// Edge messages + scatter-add: agg[dst] += relu(h[src] + bond_emb(e)).
__global__ void scatter_msg(const int* __restrict__ ei, const int* __restrict__ ea,
                            const float* __restrict__ bemb, const float* __restrict__ h,
                            float* __restrict__ agg) {
    __shared__ int sidx[5];
    int e = blockIdx.x, c = threadIdx.x;
    if (c < 2)            sidx[c] = ei[(size_t)c * NEDGES + e];   // src, dst
    else if (c < 5)       sidx[c] = ea[(size_t)e * 3 + (c - 2)];  // 3 bond features
    __syncthreads();
    if (c >= HDIM) return;
    int src = sidx[0], dst = sidx[1];
    float ev = bemb[((size_t)0 * 16 + sidx[2]) * HDIM + c]
             + bemb[((size_t)1 * 16 + sidx[3]) * HDIM + c]
             + bemb[((size_t)2 * 16 + sidx[4]) * HDIM + c];
    float m = fmaxf(h[(size_t)src * HP + c] + ev, 0.0f);
    unsafeAtomicAdd(&agg[(size_t)dst * HP + c], m);
}

// ---------------------------------------------------------------------------
// GEMM1: z = relu(BN( ((1+eps)h + agg) @ W1 + b1 )) -> bf16 [N][608]
// One block (8 waves) per 16-row M-tile. A tile staged in LDS once per K-step
// and reused by all waves; wave w owns N-tiles w, w+8, ..., (5 accumulators).
__global__ void __launch_bounds__(256)
gin_gemm1(const float* __restrict__ h, const float* __restrict__ agg,
          const float* __restrict__ epsv,
          const float* __restrict__ b1, const float* __restrict__ gamma,
          const float* __restrict__ beta, const float* __restrict__ mean,
          const float* __restrict__ var,
          const unsigned short* __restrict__ W1t,
          unsigned short* __restrict__ z) {
    __shared__ unsigned short As[16 * 32];
    const unsigned* AsU = (const unsigned*)As;
    const unsigned* WtU = (const unsigned*)W1t;

    int mtile = blockIdx.x;
    int wave  = threadIdx.x >> 5;             // 0..7
    int lane  = threadIdx.x & 31;
    int row0  = mtile * 16;
    int hi    = lane >> 4;                    // lane half
    int arow  = lane & 15;
    int koff  = hi ? 8 : 0;                   // A pair offset (K 16..31 half)
    int kb8   = hi ? 8 : 0;                   // B pair offset (K 16..31 half)
    int nl    = lane & 15;
    float epsl = 1.0f + epsv[0];

    v8f acc[5];
    #pragma unroll
    for (int t = 0; t < 5; ++t) acc[t] = (v8f){};

    for (int ks = 0; ks < KP1 / 32; ++ks) {
        __syncthreads();
        #pragma unroll
        for (int e = threadIdx.x; e < 16 * 32; e += 256) {
            int r = e >> 5, kk = e & 31, k = ks * 32 + kk;
            float vA = 0.0f;
            if (k < HDIM) {
                size_t idx = (size_t)(row0 + r) * HP + k;
                vA = epsl * h[idx] + agg[idx];
            }
            As[e] = f2bf(vA);
        }
        __syncthreads();

        Frag a;
        #pragma unroll
        for (int v = 0; v < 8; ++v) {
            int ap = v + koff;
            a.u[v] = AsU[arow * 16 + ap];
        }
        #pragma unroll
        for (int t = 0; t < 5; ++t) {
            int ntile = wave + 8 * t;
            if (ntile < NT1) {
                size_t bbase = (size_t)(ntile * 16 + nl) * (KP1 / 2) + ks * 16 + kb8;
                if (ks + 1 < KP1 / 32)
                    __builtin_prefetch(&WtU[bbase + 16], 0, 1);
                Frag b;
                #pragma unroll
                for (int v = 0; v < 8; ++v) b.u[v] = WtU[bbase + v];
                acc[t] = __builtin_amdgcn_wmma_f32_16x16x32_bf16(
                             false, a.v, false, b.v, (short)0, acc[t], false, false);
            }
        }
    }

    int mb = row0 + (hi ? 8 : 0);
    #pragma unroll
    for (int t = 0; t < 5; ++t) {
        int ntile = wave + 8 * t;
        if (ntile >= NT1) continue;
        int col = ntile * 16 + nl;
        float bb = 0.f, gm = 1.f, bt = 0.f, mn = 0.f;
        if (col < H2) {
            bb = b1[col];
            gm = gamma[col] * rsqrtf(var[col] + 1e-5f);
            bt = beta[col];
            mn = mean[col];
        }
        #pragma unroll
        for (int r = 0; r < 8; ++r) {
            float vz = acc[t][r] + bb;
            vz = (vz - mn) * gm + bt;
            vz = fmaxf(vz, 0.0f);
            if (col >= H2) vz = 0.0f;
            z[(size_t)(mb + r) * ZP + col] = f2bf(vz);
        }
    }
}

// ---------------------------------------------------------------------------
// GEMM2: h = z @ W2 + b2 -> f32 [N][304] (pad cols written 0)
// Same blocking: 8 waves, A tile in LDS once per K-step, 3 N-tiles per wave.
__global__ void __launch_bounds__(256)
gin_gemm2(const unsigned short* __restrict__ z,
          const float* __restrict__ b2,
          const unsigned short* __restrict__ W2t,
          float* __restrict__ hout) {
    __shared__ unsigned short As[16 * 32];
    const unsigned* AsU = (const unsigned*)As;
    const unsigned* WtU = (const unsigned*)W2t;

    int mtile = blockIdx.x;
    int wave  = threadIdx.x >> 5;
    int lane  = threadIdx.x & 31;
    int row0  = mtile * 16;
    int hi    = lane >> 4;
    int arow  = lane & 15;
    int koff  = hi ? 8 : 0;
    int kb8   = hi ? 8 : 0;
    int nl    = lane & 15;

    v8f acc[3];
    #pragma unroll
    for (int t = 0; t < 3; ++t) acc[t] = (v8f){};

    for (int ks = 0; ks < ZP / 32; ++ks) {
        __syncthreads();
        #pragma unroll
        for (int e = threadIdx.x; e < 16 * 32; e += 256) {
            int r = e >> 5, kk = e & 31;
            As[e] = z[(size_t)(row0 + r) * ZP + ks * 32 + kk];
        }
        __syncthreads();

        Frag a;
        #pragma unroll
        for (int v = 0; v < 8; ++v) {
            int ap = v + koff;
            a.u[v] = AsU[arow * 16 + ap];
        }
        #pragma unroll
        for (int t = 0; t < 3; ++t) {
            int ntile = wave + 8 * t;
            if (ntile < NT2) {
                size_t bbase = (size_t)(ntile * 16 + nl) * (ZP / 2) + ks * 16 + kb8;
                if (ks + 1 < ZP / 32)
                    __builtin_prefetch(&WtU[bbase + 16], 0, 1);
                Frag b;
                #pragma unroll
                for (int v = 0; v < 8; ++v) b.u[v] = WtU[bbase + v];
                acc[t] = __builtin_amdgcn_wmma_f32_16x16x32_bf16(
                             false, a.v, false, b.v, (short)0, acc[t], false, false);
            }
        }
    }

    int mb = row0 + (hi ? 8 : 0);
    #pragma unroll
    for (int t = 0; t < 3; ++t) {
        int ntile = wave + 8 * t;
        if (ntile >= NT2) continue;
        int col = ntile * 16 + nl;
        float bb = (col < HDIM) ? b2[col] : 0.0f;
        #pragma unroll
        for (int r = 0; r < 8; ++r) {
            float v = acc[t][r] + bb;
            hout[(size_t)(mb + r) * HP + col] = (col < HDIM) ? v : 0.0f;
        }
    }
}

// ---------------------------------------------------------------------------
// Mean-pool accumulation
__global__ void pool_sum(const float* __restrict__ h, const int* __restrict__ batch,
                         float* __restrict__ psum, float* __restrict__ pcnt) {
    int n = blockIdx.x, c = threadIdx.x;
    int g = batch[n];
    if (c < HDIM) unsafeAtomicAdd(&psum[(size_t)g * HP + c], h[(size_t)n * HP + c]);
    if (c == 0)   unsafeAtomicAdd(&pcnt[g], 1.0f);
}

// Final: gf = sums/max(cnt,1); out = [logits (4096x2) | graph_feature (4096x300)]
__global__ void finalize_k(const float* __restrict__ psum, const float* __restrict__ pcnt,
                           const float* __restrict__ clsW, const float* __restrict__ clsb,
                           float* __restrict__ out) {
    __shared__ float s0[320], s1[320];
    int g = blockIdx.x, c = threadIdx.x;
    float cnt = fmaxf(pcnt[g], 1.0f);
    float gf = 0.0f;
    if (c < HDIM) {
        gf = psum[(size_t)g * HP + c] / cnt;
        out[(size_t)NGRAPH * 2 + (size_t)g * HDIM + c] = gf;
    }
    s0[c] = (c < HDIM) ? gf * clsW[c * 2 + 0] : 0.0f;
    s1[c] = (c < HDIM) ? gf * clsW[c * 2 + 1] : 0.0f;
    __syncthreads();
    if (c == 0) {
        float a = 0.0f, b = 0.0f;
        for (int i = 0; i < HDIM; ++i) { a += s0[i]; b += s1[i]; }
        out[(size_t)g * 2 + 0] = a + clsb[0];
        out[(size_t)g * 2 + 1] = b + clsb[1];
    }
}

// ---------------------------------------------------------------------------
extern "C" void kernel_launch(void* const* d_in, const int* in_sizes, int n_in,
                              void* d_out, int out_size, void* d_ws, size_t ws_size,
                              hipStream_t stream) {
    (void)in_sizes; (void)n_in; (void)out_size; (void)ws_size;
    const int*   x     = (const int*)  d_in[0];
    const int*   ei    = (const int*)  d_in[1];
    const int*   ea    = (const int*)  d_in[2];
    const int*   batch = (const int*)  d_in[3];
    const float* aemb  = (const float*)d_in[4];
    const float* bemb  = (const float*)d_in[5];
    const float* eps   = (const float*)d_in[6];
    const float* W1    = (const float*)d_in[7];
    const float* b1    = (const float*)d_in[8];
    const float* gam   = (const float*)d_in[9];
    const float* bet   = (const float*)d_in[10];
    const float* mea   = (const float*)d_in[11];
    const float* var   = (const float*)d_in[12];
    const float* W2    = (const float*)d_in[13];
    const float* b2    = (const float*)d_in[14];
    const float* clsW  = (const float*)d_in[15];
    const float* clsb  = (const float*)d_in[16];
    float* out = (float*)d_out;

    char* ws = (char*)d_ws;
    size_t off = 0;
    auto carve = [&](size_t bytes) -> void* {
        void* p = ws + off;
        off += (bytes + 255) & ~(size_t)255;
        return p;
    };
    float*          h    = (float*)carve((size_t)NNODES * HP * 4);
    float*          agg  = (float*)carve((size_t)NNODES * HP * 4);
    unsigned short* z    = (unsigned short*)carve((size_t)NNODES * ZP * 2);
    unsigned short* W1t  = (unsigned short*)carve((size_t)NLAYER * ZP * KP1 * 2);
    unsigned short* W2t  = (unsigned short*)carve((size_t)NLAYER * HP * ZP * 2);
    float*          psum = (float*)carve((size_t)NGRAPH * HP * 4);
    float*          pcnt = (float*)carve((size_t)NGRAPH * 4);

    prep_weights<<<2048, 256, 0, stream>>>(W1, W2, W1t, W2t);
    atom_enc<<<NNODES, 320, 0, stream>>>(x, aemb, h);

    for (int l = 0; l < NLAYER; ++l) {
        hipMemsetAsync(agg, 0, (size_t)NNODES * HP * 4, stream);
        scatter_msg<<<NEDGES, 320, 0, stream>>>(
            ei, ea, bemb + (size_t)l * 3 * 16 * HDIM, h, agg);
        gin_gemm1<<<NNODES / 16, 256, 0, stream>>>(
            h, agg, eps + l,
            b1 + (size_t)l * H2, gam + (size_t)l * H2, bet + (size_t)l * H2,
            mea + (size_t)l * H2, var + (size_t)l * H2,
            W1t + (size_t)l * ZP * KP1, z);
        gin_gemm2<<<NNODES / 16, 256, 0, stream>>>(
            z, b2 + (size_t)l * HDIM, W2t + (size_t)l * HP * ZP, h);
    }

    hipMemsetAsync(psum, 0, (size_t)NGRAPH * HP * 4, stream);
    hipMemsetAsync(pcnt, 0, (size_t)NGRAPH * 4, stream);
    pool_sum<<<NNODES, 320, 0, stream>>>(h, batch, psum, pcnt);
    finalize_k<<<NGRAPH, 320, 0, stream>>>(psum, pcnt, clsW, clsb, out);
}
